// CrliDiscriminator_13091060318240
// MI455X (gfx1250) — matrix operand order repaired and checked by reading, not compiled, not run on hardware
//
#include <hip/hip_runtime.h>
#include <hip/hip_bf16.h>

typedef __attribute__((ext_vector_type(16))) _Float16 v16h;
typedef __attribute__((ext_vector_type(8)))  _Float16 v8h;
typedef __attribute__((ext_vector_type(8)))  float    v8f;

#define NLY 5
#define TT  256
#define DD  64
#define WAVES 2
#define RPW 16                    // batch rows per wave (WMMA M dim)
#define RPB (WAVES*RPW)           // rows per block
#define BLOCK (WAVES*32)

struct KParams {
    const float* X; const float* M; const float* L;
    const float* Wih[NLY]; const float* Whh[NLY];
    const float* bih[NLY]; const float* bhh[NLY];
    const float* Wout; const float* bout;
    float* out;
};

__device__ __forceinline__ float sigmf(float x) {
    return 1.0f / (1.0f + __expf(-x));
}
__device__ __forceinline__ float tanh_fast(float x) {
    // clamp to keep exp in range; tanh saturates well before ±10
    x = fminf(fmaxf(x, -10.0f), 10.0f);
    float e = __expf(2.0f * x);
    return (e - 1.0f) / (e + 1.0f);
}

__device__ __forceinline__ v16h mk16(v8h lo, v8h hi) {
    return __builtin_shufflevector(lo, hi, 0,1,2,3,4,5,6,7,8,9,10,11,12,13,14,15);
}

__global__ __launch_bounds__(BLOCK)
void crli_lstm_kernel(KParams p) {
    // layer geometry (constexpr -> fully unrolled layer loop)
    constexpr int H [NLY] = {32,16, 8,16,32};
    constexpr int IN[NLY] = {64,32,16, 8,16};
    constexpr int KP[NLY] = {96,64,32,32,64};     // in+H padded to mult of 32 (WMMA K)
    constexpr int G4[NLY] = {128,64,32,64,128};   // 4H gate width
    constexpr int WOFF[NLY] = {0,12288,16384,17408,19456};  // halves into sW
    constexpr int AOFF[NLY] = {0, 1536, 2560, 3072, 3584};  // halves into per-wave act
    constexpr int BO  [NLY] = {0,  128,  192,  224,  288};  // floats into sBias
    constexpr int CO  [NLY] = {0,  512,  768,  896, 1152};  // floats into per-wave c

    // packed fused weights [4H][KP] f16 (row n contiguous over k == WMMA B layout)
    __shared__ __align__(32) _Float16 sW[27648];
    __shared__ __align__(32) _Float16 sWout[64*32];
    __shared__ float sBias[416];
    __shared__ float sBout[64];
    // per-wave state
    __shared__ __align__(32) _Float16 sAct [WAVES][4608];   // [x|h] f16, K-padded, per layer
    __shared__ __align__(32) _Float16 sHout[WAVES][16*32];  // layer-4 h as its own A operand
    __shared__ float sC[WAVES][1664];                       // cell states fp32
    __shared__ float sG[WAVES][16*128];                     // gate scratch (stride 128)

    const int tid = threadIdx.x;

    // ---- pack weights: Wcat[n][k] = k<in ? Wih[n][k] : k<in+h ? Whh[n][k-in] : 0
    #pragma unroll
    for (int l = 0; l < NLY; ++l) {
        const int kp = KP[l], in = IN[l], h = H[l], g4 = G4[l];
        for (int idx = tid; idx < g4 * kp; idx += BLOCK) {
            int n = idx / kp, k = idx - n * kp;
            float v = 0.0f;
            if (k < in)          v = p.Wih[l][n * in + k];
            else if (k < in + h) v = p.Whh[l][n * h + (k - in)];
            sW[WOFF[l] + idx] = (_Float16)v;
        }
        for (int n = tid; n < g4; n += BLOCK)
            sBias[BO[l] + n] = p.bih[l][n] + p.bhh[l][n];
    }
    for (int idx = tid; idx < 64 * 32; idx += BLOCK) sWout[idx] = (_Float16)p.Wout[idx];
    for (int idx = tid; idx < 64;      idx += BLOCK) sBout[idx] = p.bout[idx];
    // zero recurrent state + K-padding once
    for (int idx = tid; idx < WAVES * 4608; idx += BLOCK) ((_Float16*)sAct)[idx] = (_Float16)0.0f;
    for (int idx = tid; idx < WAVES * 1664; idx += BLOCK) ((float*)sC)[idx] = 0.0f;
    __syncthreads();

    const int wv   = tid >> 5;
    const int lane = tid & 31;
    const int m16  = lane & 15;       // column for B/C, row for A
    const int hiw  = lane >> 4;       // half-wave select (ISA fragment striping)
    const int rowbase = blockIdx.x * RPB + wv * RPW;

    _Float16* act  = sAct[wv];
    _Float16* hout = sHout[wv];
    float*    cst  = sC[wv];
    float*    gts  = sG[wv];

    for (int t = 0; t < TT; ++t) {
        // ---- impute x_t = X*m + L*(1-m) -> f16 into layer0 input slot
        for (int i = lane; i < RPW * DD; i += 32) {
            int r = i >> 6, d = i & 63;
            size_t g = ((size_t)(rowbase + r) * TT + t) * DD + d;
            float xv = p.X[g], mv = p.M[g], lv = p.L[g];
            act[r * KP[0] + d] = (_Float16)(xv * mv + lv * (1.0f - mv));
        }
        if (t + 1 < TT) {   // hide next-step HBM latency behind WMMA work
            size_t g = ((size_t)(rowbase + m16) * TT + (t + 1)) * DD;
            if (hiw == 0) { __builtin_prefetch(&p.X[g], 0, 0); __builtin_prefetch(&p.M[g], 0, 0); }
            else          { __builtin_prefetch(&p.L[g], 0, 0); }
        }

        // ---- 5 LSTM layers, gates via V_WMMA_F32_16X16X32_F16
        #pragma unroll
        for (int l = 0; l < NLY; ++l) {
            const int kp = KP[l], in = IN[l], h = H[l];
            const int nk = KP[l] / 32, nt = G4[l] / 16;
            const _Float16* al = act + AOFF[l];

            // A fragments: 16x32 f16; lanes0-15 K=kb..kb+15 kb=0; lanes16-31 kb=16
            v16h afr[3];
            #pragma unroll
            for (int kf = 0; kf < nk; ++kf) {
                const _Float16* pa = al + m16 * kp + kf * 32 + hiw * 16;
                afr[kf] = mk16(*(const v8h*)pa, *(const v8h*)(pa + 8));
            }
            #pragma unroll
            for (int n = 0; n < nt; ++n) {
                float bv = sBias[BO[l] + n * 16 + m16];
                v8f c = {bv, bv, bv, bv, bv, bv, bv, bv};
                #pragma unroll
                for (int kf = 0; kf < nk; ++kf) {
                    // B fragment: column n=lane&15, 16 contiguous halves at K=kf*32+hiw*16
                    const _Float16* pb = sW + WOFF[l] + (n * 16 + m16) * kp + kf * 32 + hiw * 16;
                    v16h b = *(const v16h*)pb;
                    c = __builtin_amdgcn_wmma_f32_16x16x32_f16(
                            false, afr[kf], false, b, (short)0, c, false, false);
                }
                // C/D layout: VGPR r -> row hiw*8+r, col m16
                #pragma unroll
                for (int r = 0; r < 8; ++r)
                    gts[(hiw * 8 + r) * 128 + n * 16 + m16] = c[r];
            }

            // ---- elementwise gate math (PyTorch order i,f,g,o), c fp32, h -> f16
            for (int i = lane; i < RPW * h; i += 32) {
                int r = i / h, j = i - r * h;
                float ig = gts[r * 128 +         j];
                float fg = gts[r * 128 + h     + j];
                float gg = gts[r * 128 + 2 * h + j];
                float og = gts[r * 128 + 3 * h + j];
                float cold = cst[CO[l] + r * h + j];
                float cn = sigmf(fg) * cold + sigmf(ig) * tanh_fast(gg);
                float hn = sigmf(og) * tanh_fast(cn);
                cst[CO[l] + r * h + j] = cn;
                _Float16 hh = (_Float16)hn;
                act[AOFF[l] + r * kp + in + j] = hh;                 // own recurrent h slot
                if (l + 1 < NLY) act[AOFF[l + 1] + r * KP[l + 1] + j] = hh;  // next layer input
                else             hout[r * 32 + j] = hh;              // feed output projection
            }
        }

        // ---- output projection: h4[16x32] @ Wout^T[32x64] + b_out, direct to HBM
        {
            const _Float16* pa = hout + m16 * 32 + hiw * 16;
            v16h a = mk16(*(const v8h*)pa, *(const v8h*)(pa + 8));
            #pragma unroll
            for (int n = 0; n < 4; ++n) {
                float bv = sBout[n * 16 + m16];
                v8f c = {bv, bv, bv, bv, bv, bv, bv, bv};
                const _Float16* pb = sWout + (n * 16 + m16) * 32 + hiw * 16;
                v16h b = *(const v16h*)pb;
                c = __builtin_amdgcn_wmma_f32_16x16x32_f16(
                        false, a, false, b, (short)0, c, false, false);
                #pragma unroll
                for (int r = 0; r < 8; ++r) {
                    int m = hiw * 8 + r;
                    p.out[((size_t)(rowbase + m) * TT + t) * DD + n * 16 + m16] = c[r];
                }
            }
        }
    }
}

extern "C" void kernel_launch(void* const* d_in, const int* in_sizes, int n_in,
                              void* d_out, int out_size, void* d_ws, size_t ws_size,
                              hipStream_t stream) {
    (void)d_ws; (void)ws_size; (void)out_size; (void)n_in;
    KParams p;
    p.X = (const float*)d_in[0];
    p.M = (const float*)d_in[1];
    p.L = (const float*)d_in[2];
    // params tuple flattened in dict order: (Wih, Whh, bih, bhh) x 5 layers
    for (int l = 0; l < NLY; ++l) {
        p.Wih[l] = (const float*)d_in[3 + 4 * l + 0];
        p.Whh[l] = (const float*)d_in[3 + 4 * l + 1];
        p.bih[l] = (const float*)d_in[3 + 4 * l + 2];
        p.bhh[l] = (const float*)d_in[3 + 4 * l + 3];
    }
    p.Wout = (const float*)d_in[23];
    p.bout = (const float*)d_in[24];
    p.out  = (float*)d_out;

    const int B = in_sizes[0] / (TT * DD);     // 2048
    dim3 grid(B / RPB), block(BLOCK);          // 64 blocks x 2 waves, 16 rows/wave
    crli_lstm_kernel<<<grid, block, 0, stream>>>(p);
}
